// MultiHeadedAttention_5617817223386
// MI455X (gfx1250) — compile-verified, matrix operand order checked by reading, not run on hardware
//
#include <hip/hip_runtime.h>
#include <hip/hip_bf16.h>
#include <math.h>

#define B_    4
#define S_    1024
#define HID_  2048
#define NH_   16
#define D_    128
#define ND_   2048
#define MTOT_ 4096
#define LS    40   // LDS row stride in bf16 elements (80B: 16B-aligned, conflict-breaking)

typedef __attribute__((ext_vector_type(16))) __bf16 v16bf;
typedef __attribute__((ext_vector_type(8)))  float  v8f;

// Load one 16x32 bf16 WMMA operand fragment from LDS.
// Lo lanes: K=0..7 (contig) then K=16..23 (contig at +16); hi lanes: K=8..15 / K=24..31.
__device__ __forceinline__ v16bf ld_frag(const __bf16* __restrict__ p) {
  v16bf f;
#pragma unroll
  for (int i = 0; i < 8; ++i) f[i] = p[i];
#pragma unroll
  for (int i = 0; i < 8; ++i) f[8 + i] = p[16 + i];
  return f;
}

#define WMMA_BF16(A, Bm, C) \
  __builtin_amdgcn_wmma_f32_16x16x32_bf16(false, (A), false, (Bm), (short)0, (C), false, false)

// pack two fp32 -> packed bf16x2 (u32)
__device__ __forceinline__ unsigned pk2(float lo, float hi) {
  union { __bf16 h[2]; unsigned u; } x;
  x.h[0] = (__bf16)lo;
  x.h[1] = (__bf16)hi;
  return x.u;
}
__device__ __forceinline__ float fsel(const float4& v, int j) {
  return (j == 0) ? v.x : (j == 1) ? v.y : (j == 2) ? v.z : v.w;
}

// --- CDNA5 async global->LDS copy (16B per lane), tracked by ASYNCcnt -------
// Generic pointers to __shared__ carry the wave-relative LDS offset in their
// low 32 bits (LDS aperture: addr[63:32]==SHARED_BASE), so truncation gives
// the VDST LDS-address operand.
__device__ __forceinline__ void async_cp16(void* lds, const void* g) {
  unsigned ldsoff = (unsigned)(size_t)lds;
  unsigned long long ga = (unsigned long long)(size_t)g;
  asm volatile("global_load_async_to_lds_b128 %0, %1, off"
               :: "v"(ldsoff), "v"(ga) : "memory");
}
__device__ __forceinline__ void wait_async0() {
  asm volatile("s_wait_asynccnt 0x0" ::: "memory");
}

// ---------------------------------------------------------------------------
// Kernel 1: QKV projection GEMM (fp32 -> bf16 WMMA, fp32 acc) + fused RoPE.
// grid = (ND/128, MTOT/64, 3)  z: 0=q, 1=k, 2=v.  Double-buffered LDS.
// q,k stored [b,h,s,d]; v stored TRANSPOSED [b,h,d,s] for async PV staging.
// ---------------------------------------------------------------------------
__global__ __launch_bounds__(256)
void qkv_rope_kernel(const float* __restrict__ Xq, const float* __restrict__ Xk,
                     const float* __restrict__ Xv,
                     const float* __restrict__ Wq, const float* __restrict__ Wk,
                     const float* __restrict__ Wv,
                     const float* __restrict__ Bq, const float* __restrict__ Bk,
                     const float* __restrict__ Bv,
                     const float* __restrict__ fc,
                     __bf16* __restrict__ qout, __bf16* __restrict__ kout,
                     __bf16* __restrict__ vtr)
{
  const int z = blockIdx.z;
  const float* X    = (z == 0) ? Xq : (z == 1) ? Xk : Xv;
  const float* W    = (z == 0) ? Wq : (z == 1) ? Wk : Wv;
  const float* bias = (z == 0) ? Bq : (z == 1) ? Bk : Bv;
  __bf16* dst       = (z == 0) ? qout : (z == 1) ? kout : vtr;

  __shared__ __align__(16) __bf16 As[2][64 * LS];
  __shared__ __align__(16) __bf16 Bs[2][128 * LS];

  const int t = threadIdx.x, lane = t & 31, wid = t >> 5;
  const int wm = wid >> 2, wn = wid & 3;
  const int ln = lane & 15, hi = lane >> 4, kb = hi << 3;
  const int m0 = blockIdx.y * 64, n0 = blockIdx.x * 128;

  const int ar = t >> 2, ac = (t & 3) * 8;          // A: 64 x 32, 8 contig per thread
  const int bn4 = (t & 31) * 4, bk4 = (t >> 5) * 4; // W: 4k x 4n sub-tile per thread

  float  a8[8];
  float4 b4[4];

  auto gload = [&](int kk) {
    const float* ap = X + (size_t)(m0 + ar) * HID_ + kk + ac;
#pragma unroll
    for (int i = 0; i < 8; ++i) a8[i] = ap[i];
    const float* bp = W + (size_t)(kk + bk4) * ND_ + n0 + bn4;
#pragma unroll
    for (int r = 0; r < 4; ++r) b4[r] = *(const float4*)(bp + (size_t)r * ND_);
  };
  auto lstore = [&](int bufi) {
    uint4 av;
    av.x = pk2(a8[0], a8[1]); av.y = pk2(a8[2], a8[3]);
    av.z = pk2(a8[4], a8[5]); av.w = pk2(a8[6], a8[7]);
    *(uint4*)&As[bufi][ar * LS + ac] = av;
#pragma unroll
    for (int j = 0; j < 4; ++j) {                    // transpose -> Bs[n][k], b64 stores
      uint2 bv;
      bv.x = pk2(fsel(b4[0], j), fsel(b4[1], j));
      bv.y = pk2(fsel(b4[2], j), fsel(b4[3], j));
      *(uint2*)&Bs[bufi][(bn4 + j) * LS + bk4] = bv;
    }
  };

  v8f acc[2][2] = {};
  gload(0);
  lstore(0);
  __syncthreads();
  const int nIt = HID_ / 32;
  for (int it = 0; it < nIt; ++it) {
    const int cur = it & 1;
    if (it + 1 < nIt) gload((it + 1) * 32);          // prefetch next tile to regs
    v16bf a0 = ld_frag(&As[cur][(wm * 32 + ln) * LS + kb]);
    v16bf a1 = ld_frag(&As[cur][(wm * 32 + 16 + ln) * LS + kb]);
    v16bf b0 = ld_frag(&Bs[cur][(wn * 32 + ln) * LS + kb]);
    v16bf b1 = ld_frag(&Bs[cur][(wn * 32 + 16 + ln) * LS + kb]);
    acc[0][0] = WMMA_BF16(a0, b0, acc[0][0]);
    acc[0][1] = WMMA_BF16(a0, b1, acc[0][1]);
    acc[1][0] = WMMA_BF16(a1, b0, acc[1][0]);
    acc[1][1] = WMMA_BF16(a1, b1, acc[1][1]);
    if (it + 1 < nIt) lstore(cur ^ 1);
    __syncthreads();
  }

#pragma unroll
  for (int i = 0; i < 2; ++i) {
#pragma unroll
    for (int j = 0; j < 2; ++j) {
      const int col = n0 + wn * 32 + j * 16 + ln;
      const int h = col >> 7;
      const int d = col & 127;
      const float bval = bias[col];
#pragma unroll
      for (int r = 0; r < 8; ++r) {
        const int row = m0 + wm * 32 + i * 16 + r + hi * 8;
        const int bb  = row >> 10;
        const int s   = row & (S_ - 1);
        float v = acc[i][j][r] + bval;
        if (z < 2) {
          // RoPE: pair (2i,2i+1) lives in adjacent lanes (col parity == lane parity)
          float p = __shfl_xor(v, 1, 32);
          const float* f2 = fc + ((size_t)s * (D_ / 2) + (d >> 1)) * 2;
          const float c = f2[0], sn = f2[1];
          v = ((d & 1) == 0) ? (v * c - p * sn) : (p * sn + v * c);
        }
        const size_t idx = (z == 2)
            ? ((((size_t)(bb * NH_ + h)) * D_ + d) * S_ + s)   // v: [b,h,d,s]
            : ((((size_t)(bb * NH_ + h)) * S_ + s) * D_ + d);  // q,k: [b,h,s,d]
        dst[idx] = (__bf16)v;
      }
    }
  }
}

// ---------------------------------------------------------------------------
// Kernel 2: scores = qk^T, fused (s+pb)/sqrt(D)+mask+prev_attn -> prev_attn_out
// grid = (S/128, S/64, B*NH). All LDS staging via async global->LDS (ASYNCcnt).
// ---------------------------------------------------------------------------
__global__ __launch_bounds__(256)
void scores_kernel(const __bf16* __restrict__ qbf, const __bf16* __restrict__ kbf,
                   const float* __restrict__ pbias, const float* __restrict__ mask,
                   const float* __restrict__ pattn, float* __restrict__ prevout)
{
  const int bh = blockIdx.z;
  const int b  = bh >> 4;
  const int h  = bh & 15;
  const int q0 = blockIdx.y * 64;
  const int c0 = blockIdx.x * 128;
  const __bf16* qb = qbf + (size_t)bh * S_ * D_;
  const __bf16* kp = kbf + (size_t)bh * S_ * D_;

  __shared__ __align__(16) __bf16 As[2][64 * LS];
  __shared__ __align__(16) __bf16 Bs[2][128 * LS];

  const int t = threadIdx.x, lane = t & 31, wid = t >> 5;
  const int wm = wid >> 2, wn = wid & 3;
  const int ln = lane & 15, hi = lane >> 4, kb = hi << 3;

  const int ar = t >> 2, ac = (t & 3) * 8;   // A: 64x32, one 16B chunk per thread
  const int br = t >> 1, bc = (t & 1) * 16;  // B: 128x32, two 16B chunks per thread

  auto gissue = [&](int kk, int bufi) {
    async_cp16(&As[bufi][ar * LS + ac], qb + (size_t)(q0 + ar) * D_ + kk + ac);
    const __bf16* bp = kp + (size_t)(c0 + br) * D_ + kk + bc;
    async_cp16(&Bs[bufi][br * LS + bc], bp);
    async_cp16(&Bs[bufi][br * LS + bc + 8], bp + 8);
  };

  v8f acc[2][2] = {};
  gissue(0, 0);
  wait_async0();
  __syncthreads();
  const int nIt = D_ / 32;
  for (int it = 0; it < nIt; ++it) {
    const int cur = it & 1;
    if (it + 1 < nIt) gissue((it + 1) * 32, cur ^ 1);  // async DMA under the WMMAs
    v16bf a0 = ld_frag(&As[cur][(wm * 32 + ln) * LS + kb]);
    v16bf a1 = ld_frag(&As[cur][(wm * 32 + 16 + ln) * LS + kb]);
    v16bf b0 = ld_frag(&Bs[cur][(wn * 32 + ln) * LS + kb]);
    v16bf b1 = ld_frag(&Bs[cur][(wn * 32 + 16 + ln) * LS + kb]);
    acc[0][0] = WMMA_BF16(a0, b0, acc[0][0]);
    acc[0][1] = WMMA_BF16(a0, b1, acc[0][1]);
    acc[1][0] = WMMA_BF16(a1, b0, acc[1][0]);
    acc[1][1] = WMMA_BF16(a1, b1, acc[1][1]);
    if (it + 1 < nIt) wait_async0();
    __syncthreads();
  }

  const float scale = 0.088388347648318447f;  // 1/sqrt(128)
#pragma unroll
  for (int i = 0; i < 2; ++i) {
#pragma unroll
    for (int j = 0; j < 2; ++j) {
      const int kg = c0 + wn * 32 + j * 16 + ln;
#pragma unroll
      for (int r = 0; r < 8; ++r) {
        const int qg = q0 + wm * 32 + i * 16 + r + hi * 8;
        const size_t qk = (size_t)qg * S_ + kg;
        float sc = (acc[i][j][r] + pbias[(size_t)h * S_ * S_ + qk]) * scale
                 + mask[(size_t)b * S_ * S_ + qk]
                 + pattn[(size_t)bh * S_ * S_ + qk];
        prevout[(size_t)bh * S_ * S_ + qk] = sc;
      }
    }
  }
}

// ---------------------------------------------------------------------------
// Kernel 3: per-row softmax stats (max, sum) over S=1024. One row per block.
// ---------------------------------------------------------------------------
__global__ __launch_bounds__(256)
void softmax_stats_kernel(const float* __restrict__ scores,
                          float* __restrict__ rmax, float* __restrict__ rsum)
{
  const int row = blockIdx.x;
  const float4 v = reinterpret_cast<const float4*>(scores + (size_t)row * S_)[threadIdx.x];
  const int t = threadIdx.x;
  __shared__ float red[256];
  float m = fmaxf(fmaxf(v.x, v.y), fmaxf(v.z, v.w));
  red[t] = m; __syncthreads();
  for (int s = 128; s > 0; s >>= 1) { if (t < s) red[t] = fmaxf(red[t], red[t + s]); __syncthreads(); }
  m = red[0]; __syncthreads();
  float sum = __expf(v.x - m) + __expf(v.y - m) + __expf(v.z - m) + __expf(v.w - m);
  red[t] = sum; __syncthreads();
  for (int s = 128; s > 0; s >>= 1) { if (t < s) red[t] += red[t + s]; __syncthreads(); }
  if (t == 0) { rmax[row] = m; rsum[row] = red[0]; }
}

// ---------------------------------------------------------------------------
// Kernel 4: ctx = softmax(scores) @ v. Probs formed on the fly (exp -> bf16,
// VALU path); v tile ([d][s] pre-transposed in ws) staged via async DMA.
// grid = (S/64, B*NH)
// ---------------------------------------------------------------------------
__global__ __launch_bounds__(256)
void ctx_kernel(const float* __restrict__ scores, const float* __restrict__ rmax,
                const float* __restrict__ rsum, const __bf16* __restrict__ vtr,
                __bf16* __restrict__ ctxbf)
{
  const int bh = blockIdx.y;
  const int b  = bh >> 4;
  const int h  = bh & 15;
  const int q0 = blockIdx.x * 64;
  const float* sp  = scores + (size_t)bh * S_ * S_;
  const __bf16* vb = vtr + (size_t)bh * D_ * S_;   // [d][s]

  __shared__ __align__(16) __bf16 As[2][64 * LS];
  __shared__ __align__(16) __bf16 Bs[2][128 * LS];

  const int t = threadIdx.x, lane = t & 31, wid = t >> 5;
  const int wm = wid >> 2, wn = wid & 3;
  const int ln = lane & 15, hi = lane >> 4, kb = hi << 3;

  const int ar = t >> 2, ac = (t & 3) * 8;
  const int vd = t >> 1, vc = (t & 1) * 16;        // B: 128d x 32s, 2 chunks/thread

  const float rm = rmax[bh * S_ + q0 + ar];
  const float ri = 1.0f / rsum[bh * S_ + q0 + ar];

  float a8[8];

  auto gload = [&](int kc) {
    const float* ap = sp + (size_t)(q0 + ar) * S_ + kc + ac;
#pragma unroll
    for (int i = 0; i < 8; ++i) a8[i] = __expf(ap[i] - rm) * ri;
  };
  auto bissue = [&](int kc, int bufi) {
    const __bf16* bp = vb + (size_t)vd * S_ + kc + vc;
    async_cp16(&Bs[bufi][vd * LS + vc], bp);
    async_cp16(&Bs[bufi][vd * LS + vc + 8], bp + 8);
  };
  auto astore = [&](int bufi) {
    uint4 av;
    av.x = pk2(a8[0], a8[1]); av.y = pk2(a8[2], a8[3]);
    av.z = pk2(a8[4], a8[5]); av.w = pk2(a8[6], a8[7]);
    *(uint4*)&As[bufi][ar * LS + ac] = av;
  };

  v8f acc[2][2] = {};
  gload(0);
  bissue(0, 0);
  astore(0);
  wait_async0();
  __syncthreads();
  const int nIt = S_ / 32;
  for (int it = 0; it < nIt; ++it) {
    const int cur = it & 1;
    if (it + 1 < nIt) { bissue((it + 1) * 32, cur ^ 1); gload((it + 1) * 32); }
    v16bf a0 = ld_frag(&As[cur][(wm * 32 + ln) * LS + kb]);
    v16bf a1 = ld_frag(&As[cur][(wm * 32 + 16 + ln) * LS + kb]);
    v16bf b0 = ld_frag(&Bs[cur][(wn * 32 + ln) * LS + kb]);
    v16bf b1 = ld_frag(&Bs[cur][(wn * 32 + 16 + ln) * LS + kb]);
    acc[0][0] = WMMA_BF16(a0, b0, acc[0][0]);
    acc[0][1] = WMMA_BF16(a0, b1, acc[0][1]);
    acc[1][0] = WMMA_BF16(a1, b0, acc[1][0]);
    acc[1][1] = WMMA_BF16(a1, b1, acc[1][1]);
    if (it + 1 < nIt) { astore(cur ^ 1); wait_async0(); }
    __syncthreads();
  }

#pragma unroll
  for (int i = 0; i < 2; ++i) {
#pragma unroll
    for (int j = 0; j < 2; ++j) {
      const int dg = wn * 32 + j * 16 + ln;
#pragma unroll
      for (int r = 0; r < 8; ++r) {
        const int qg = q0 + wm * 32 + i * 16 + r + hi * 8;
        ctxbf[((size_t)(b * S_) + qg) * ND_ + h * D_ + dg] = (__bf16)acc[i][j][r];
      }
    }
  }
}

// ---------------------------------------------------------------------------
// Kernel 5: out = ctx @ wo + bo  (4096 x 2048 x 2048). A tile via async DMA,
// B tile via fp32->bf16 register path. grid = (HID/128, MTOT/64)
// ---------------------------------------------------------------------------
__global__ __launch_bounds__(256)
void out_gemm_kernel(const __bf16* __restrict__ ctxbf, const float* __restrict__ wo,
                     const float* __restrict__ bo, float* __restrict__ out)
{
  __shared__ __align__(16) __bf16 As[2][64 * LS];
  __shared__ __align__(16) __bf16 Bs[2][128 * LS];

  const int t = threadIdx.x, lane = t & 31, wid = t >> 5;
  const int wm = wid >> 2, wn = wid & 3;
  const int ln = lane & 15, hi = lane >> 4, kb = hi << 3;

  const int m0 = blockIdx.y * 64;
  const int n0 = blockIdx.x * 128;

  const int ar = t >> 2, ac = (t & 3) * 8;
  const int bn4 = (t & 31) * 4, bk4 = (t >> 5) * 4;

  float4 b4[4];

  auto aissue = [&](int kk, int bufi) {
    async_cp16(&As[bufi][ar * LS + ac], ctxbf + (size_t)(m0 + ar) * ND_ + kk + ac);
  };
  auto bload = [&](int kk) {
    const float* bp = wo + (size_t)(kk + bk4) * HID_ + n0 + bn4;
#pragma unroll
    for (int r = 0; r < 4; ++r) b4[r] = *(const float4*)(bp + (size_t)r * HID_);
  };
  auto bstore = [&](int bufi) {
#pragma unroll
    for (int j = 0; j < 4; ++j) {
      uint2 bv;
      bv.x = pk2(fsel(b4[0], j), fsel(b4[1], j));
      bv.y = pk2(fsel(b4[2], j), fsel(b4[3], j));
      *(uint2*)&Bs[bufi][(bn4 + j) * LS + bk4] = bv;
    }
  };

  v8f acc[2][2] = {};
  aissue(0, 0);
  bload(0);
  bstore(0);
  wait_async0();
  __syncthreads();
  const int nIt = ND_ / 32;
  for (int it = 0; it < nIt; ++it) {
    const int cur = it & 1;
    if (it + 1 < nIt) { aissue((it + 1) * 32, cur ^ 1); bload((it + 1) * 32); }
    v16bf a0 = ld_frag(&As[cur][(wm * 32 + ln) * LS + kb]);
    v16bf a1 = ld_frag(&As[cur][(wm * 32 + 16 + ln) * LS + kb]);
    v16bf b0 = ld_frag(&Bs[cur][(wn * 32 + ln) * LS + kb]);
    v16bf b1 = ld_frag(&Bs[cur][(wn * 32 + 16 + ln) * LS + kb]);
    acc[0][0] = WMMA_BF16(a0, b0, acc[0][0]);
    acc[0][1] = WMMA_BF16(a0, b1, acc[0][1]);
    acc[1][0] = WMMA_BF16(a1, b0, acc[1][0]);
    acc[1][1] = WMMA_BF16(a1, b1, acc[1][1]);
    if (it + 1 < nIt) { bstore(cur ^ 1); wait_async0(); }
    __syncthreads();
  }

#pragma unroll
  for (int i = 0; i < 2; ++i) {
#pragma unroll
    for (int j = 0; j < 2; ++j) {
      const int col = n0 + wn * 32 + j * 16 + ln;
      const float bval = bo[col];
#pragma unroll
      for (int r = 0; r < 8; ++r) {
        const int row = m0 + wm * 32 + i * 16 + r + hi * 8;
        out[(size_t)row * HID_ + col] = acc[i][j][r] + bval;
      }
    }
  }
}

// ---------------------------------------------------------------------------
extern "C" void kernel_launch(void* const* d_in, const int* in_sizes, int n_in,
                              void* d_out, int out_size, void* d_ws, size_t ws_size,
                              hipStream_t stream)
{
  const float* key   = (const float*)d_in[0];
  const float* value = (const float*)d_in[1];
  const float* query = (const float*)d_in[2];
  const float* mask  = (const float*)d_in[3];
  const float* pbias = (const float*)d_in[4];
  const float* pattn = (const float*)d_in[5];
  const float* fc    = (const float*)d_in[6];
  const float* wq    = (const float*)d_in[7];
  const float* bq    = (const float*)d_in[8];
  const float* wk    = (const float*)d_in[9];
  const float* bk    = (const float*)d_in[10];
  const float* wv    = (const float*)d_in[11];
  const float* bv    = (const float*)d_in[12];
  const float* wo    = (const float*)d_in[13];
  const float* bo    = (const float*)d_in[14];

  float* out     = (float*)d_out;                       // B*S*HID
  float* prevout = out + (size_t)B_ * S_ * HID_;        // B*NH*S*S

  char* ws = (char*)d_ws;
  __bf16* qbf   = (__bf16*)(ws);                        // 16 MB  [b,h,s,d]
  __bf16* kbf   = (__bf16*)(ws + (16ull << 20));        // 16 MB  [b,h,s,d]
  __bf16* vtr   = (__bf16*)(ws + (32ull << 20));        // 16 MB  [b,h,d,s]
  __bf16* ctxbf = (__bf16*)(ws + (48ull << 20));        // 16 MB  [b,s,h*d]
  float*  rmax  = (float*)(ws + (64ull << 20));         // 256 KB
  float*  rsum  = (float*)(ws + (64ull << 20) + (size_t)B_ * NH_ * S_ * sizeof(float));

  qkv_rope_kernel<<<dim3(ND_ / 128, MTOT_ / 64, 3), 256, 0, stream>>>(
      query, key, value, wq, wk, wv, bq, bk, bv, fc, qbf, kbf, vtr);

  scores_kernel<<<dim3(S_ / 128, S_ / 64, B_ * NH_), 256, 0, stream>>>(
      qbf, kbf, pbias, mask, pattn, prevout);

  softmax_stats_kernel<<<dim3(B_ * NH_ * S_), 256, 0, stream>>>(prevout, rmax, rsum);

  ctx_kernel<<<dim3(S_ / 64, B_ * NH_), 256, 0, stream>>>(prevout, rmax, rsum, vtr, ctxbf);

  out_gemm_kernel<<<dim3(HID_ / 128, MTOT_ / 64), 256, 0, stream>>>(ctxbf, wo, bo, out);
}
